// CausalGroupQueryAttention_28827820490879
// MI455X (gfx1250) — compile-verified, matrix operand order checked by reading, not run on hardware
//
#include <hip/hip_runtime.h>

// ---------------------------------------------------------------------------
// CausalGroupQueryAttention on MI455X (gfx1250, wave32, WMMA bf16 + TDM)
//   B=2, T=2048, C=2048, H=16, Hkv=4, D=128
// Pipeline: fp32->bf16 convert, N-major weight transpose, WMMA GEMMs with
// TDM-staged (tensor_load_to_lds) double-buffered A tiles, flash attention
// with online softmax, WMMA output projection.
// ---------------------------------------------------------------------------

typedef __attribute__((ext_vector_type(16))) __bf16        v16bf;
typedef __attribute__((ext_vector_type(8)))  float         v8f;
typedef __attribute__((ext_vector_type(4)))  unsigned int  u32x4;
typedef __attribute__((ext_vector_type(8)))  unsigned int  u32x8;

#define B_SZ   2
#define T_SZ   2048
#define C_SZ   2048
#define NH     16
#define NKV    4
#define HD     128
#define KVN    1024          // 2 * HD * NKV
#define NEG_INF (-1e30f)

__device__ __forceinline__ unsigned short f32_bf16(float f) {
    unsigned int u = __float_as_uint(f);
    u += 0x7FFFu + ((u >> 16) & 1u);   // round-to-nearest-even
    return (unsigned short)(u >> 16);
}

// Load one 16x32 bf16 A/B fragment for this lane: 8 contiguous bf16 at p,
// 8 more at p+16 (elements). Two b128 loads per lane (global or ds).
__device__ __forceinline__ v16bf load_frag(const unsigned short* p) {
    union { uint4 u[2]; v16bf v; } f;
    f.u[0] = *reinterpret_cast<const uint4*>(p);
    f.u[1] = *reinterpret_cast<const uint4*>(p + 16);
    return f.v;
}

// ---------------------------------------------------------------------------
// Elementwise converts / transposes (memory-bound prep; all data ends in L2)
// ---------------------------------------------------------------------------
__global__ void cvt_f32_bf16_kernel(const float* __restrict__ in,
                                    unsigned short* __restrict__ out, size_t n) {
    size_t i = (size_t)blockIdx.x * blockDim.x + threadIdx.x;
    if (i < n) out[i] = f32_bf16(in[i]);
}

// in: [K,N] fp32 row-major  ->  out: [N,K] bf16 row-major (B^T for WMMA)
__global__ void transpose_w_kernel(const float* __restrict__ in,
                                   unsigned short* __restrict__ out,
                                   int K, int N) {
    size_t i = (size_t)blockIdx.x * blockDim.x + threadIdx.x;
    if (i >= (size_t)K * N) return;
    int n = (int)(i / K);
    int k = (int)(i % K);
    out[i] = f32_bf16(in[(size_t)k * N + n]);
}

// kvf: [B*T, KVN] bf16 -> vt: [B*NKV, HD, T] bf16   (V transposed, t-contiguous)
__global__ void vt_transpose_kernel(const unsigned short* __restrict__ kvf,
                                    unsigned short* __restrict__ vt) {
    size_t i = (size_t)blockIdx.x * blockDim.x + threadIdx.x;   // < 8*128*2048
    int t  = (int)(i & (T_SZ - 1));
    int d  = (int)((i >> 11) & (HD - 1));
    int bh = (int)(i >> 18);
    int b = bh >> 2, hkv = bh & 3;
    vt[i] = kvf[(size_t)(b * T_SZ + t) * KVN + (NKV * HD) + hkv * HD + d];
}

// ---------------------------------------------------------------------------
// Issue one 2D TDM tile load: 64 rows x 32 cols of bf16 from a row-major
// [M,K] matrix into LDS (row-major 64x32). ISA 8.3/8.4 descriptor layout.
// ---------------------------------------------------------------------------
__device__ __forceinline__ void tdm_load_tile_64x32(
    unsigned long long gaddr, unsigned lds_addr, const u32x8& g1) {
    u32x4 g0;
    g0[0] = 1u;                                   // count=1, user mode
    g0[1] = lds_addr;                             // LDS byte address
    g0[2] = (unsigned)gaddr;                      // global_addr[31:0]
    g0[3] = (unsigned)((gaddr >> 32) & 0x1FFFFFFull) | (2u << 30); // [56:32]|type=2
    asm volatile("tensor_load_to_lds %0, %1" :: "s"(g0), "s"(g1) : "memory");
}

// ---------------------------------------------------------------------------
// Generic WMMA GEMM: Out[M,N] = A[M,K] * B[K,N], with Bt = B^T ([N,K] bf16).
// Workgroup = 4 waves, tile 64x128; each wave: 4 M-tiles x 2 N-tiles (64x32).
// A tile (64x32, shared by all 4 waves) is DMA'd by the Tensor Data Mover
// into double-buffered LDS; B fragments stream from global (L2-resident).
// ---------------------------------------------------------------------------
template <bool OUT_BF16>
__global__ __launch_bounds__(128) void gemm_bf16_kernel(
    const unsigned short* __restrict__ A,
    const unsigned short* __restrict__ Bt,
    void* __restrict__ Out, int M, int N, int K)
{
    __shared__ unsigned short Abuf[2][64 * 32];   // 2 x 4KB double buffer

    const int lane = threadIdx.x & 31;
    const int w    = threadIdx.x >> 5;
    const int l15  = lane & 15;
    const int hi   = lane >> 4;
    const int kb8  = hi * 8;
    const int bm   = blockIdx.x * 64;
    const int bn   = blockIdx.y * 128 + w * 32;

    v8f acc[4][2];
    for (int mt = 0; mt < 4; mt++)
        for (int nt = 0; nt < 2; nt++)
            for (int r = 0; r < 8; r++) acc[mt][nt][r] = 0.f;

    const unsigned short* Brow[2];
    for (int nt = 0; nt < 2; nt++)
        Brow[nt] = Bt + (size_t)(bn + nt * 16 + l15) * K + kb8;

    // --- TDM descriptor group 1 (uniform across the whole kernel) ---
    u32x8 g1;
    g1[0] = 0x00010000u;                                  // data_size = 2 bytes
    g1[1] = ((unsigned)K & 0xFFFFu) << 16;                // tensor_dim0[15:0]
    g1[2] = (((unsigned)M & 0xFFFFu) << 16) | ((unsigned)K >> 16); // dim1|dim0_hi
    g1[3] = (32u << 16) | (((unsigned)M >> 16) & 0xFFFFu);// tile_dim0=32 | dim1_hi
    g1[4] = 64u;                                          // tile_dim1=64, dim2=0
    g1[5] = (unsigned)K;                                  // dim0_stride[31:0]
    g1[6] = 0u;
    g1[7] = 0u;

    const unsigned lds0 = (unsigned)(size_t)(&Abuf[0][0]);
    const unsigned lds1 = (unsigned)(size_t)(&Abuf[1][0]);
    const unsigned long long gbase =
        (unsigned long long)(size_t)A + (unsigned long long)bm * K * 2ull;

    const int niter = K / 32;

    if (w == 0) tdm_load_tile_64x32(gbase, lds0, g1);     // preload tile 0

    for (int i = 0; i < niter; i++) {
        if (w == 0) {
            if (i + 1 < niter) {   // issue next tile into the other buffer
                tdm_load_tile_64x32(gbase + (unsigned long long)(i + 1) * 64ull,
                                    ((i + 1) & 1) ? lds1 : lds0, g1);
                __builtin_amdgcn_s_wait_tensorcnt(1);     // tile i complete
            } else {
                __builtin_amdgcn_s_wait_tensorcnt(0);
            }
        }
        __syncthreads();                                  // tile i visible to all

        const unsigned short* Ab = Abuf[i & 1];
        v16bf a[4], b[2];
        for (int mt = 0; mt < 4; mt++)
            a[mt] = load_frag(Ab + (mt * 16 + l15) * 32 + kb8);   // ds_load_b128
        const int k = i * 32;
        for (int nt = 0; nt < 2; nt++) b[nt] = load_frag(Brow[nt] + k);
        if (i + 1 < niter) __builtin_prefetch(Brow[0] + k + 32, 0, 1);

        asm volatile("s_wait_dscnt 0x0" ::: "memory");    // frags in regs
        __syncthreads();                                  // safe to overwrite

        for (int mt = 0; mt < 4; mt++)
            for (int nt = 0; nt < 2; nt++)
                acc[mt][nt] = __builtin_amdgcn_wmma_f32_16x16x32_bf16(
                    false, a[mt], false, b[nt], (short)0, acc[mt][nt], false, false);
    }

    for (int mt = 0; mt < 4; mt++)
        for (int nt = 0; nt < 2; nt++)
            for (int r = 0; r < 8; r++) {
                int row = bm + mt * 16 + r + 8 * hi;
                int col = bn + nt * 16 + l15;
                float v = acc[mt][nt][r];
                if (OUT_BF16)
                    ((unsigned short*)Out)[(size_t)row * N + col] = f32_bf16(v);
                else
                    ((float*)Out)[(size_t)row * N + col] = v;
            }
}

// ---------------------------------------------------------------------------
// Flash attention: grid (T/64, B*H), 4 waves/WG. Wave owns 16 q rows, D=128.
// 32-key blocks: S = Q K^T (4 wmma/tile), online softmax, P via LDS re-layout,
// O += P V^T (8 wmma). Loop bound uniform per WG -> EXEC all-ones for WMMA.
// ---------------------------------------------------------------------------
__global__ __launch_bounds__(128) void attn_kernel(
    const unsigned short* __restrict__ Qf,   // [B*T, C] bf16
    const unsigned short* __restrict__ KVf,  // [B*T, KVN] bf16 (K in cols 0..511)
    const unsigned short* __restrict__ Vt,   // [B*NKV, HD, T] bf16
    unsigned short* __restrict__ Yb)         // [B*T, C] bf16
{
    __shared__ unsigned short Plds[4][16 * 32];   // per-wave P tile

    const int lane = threadIdx.x & 31;
    const int w    = threadIdx.x >> 5;
    const int l15  = lane & 15;
    const int hi   = lane >> 4;
    const int kb8  = hi * 8;

    const int qblk = blockIdx.x;          // 0..31 (64-row q block)
    const int bh   = blockIdx.y;          // 0..31
    const int b    = bh >> 4;
    const int h    = bh & 15;
    const int hkv  = h >> 2;              // GQA: 4 q-heads per kv-head
    const int q0   = qblk * 64 + w * 16;  // this wave's q-tile base

    // Q fragments: 4 d-steps of 32
    v16bf qfrag[4];
    {
        const unsigned short* qp =
            Qf + (size_t)(b * T_SZ + q0 + l15) * C_SZ + h * HD + kb8;
        for (int ds = 0; ds < 4; ds++) qfrag[ds] = load_frag(qp + ds * 32);
    }

    float mrow[8], lrow[8];
    v8f O[8];
    for (int r = 0; r < 8; r++) { mrow[r] = NEG_INF; lrow[r] = 0.f; }
    for (int dt = 0; dt < 8; dt++)
        for (int r = 0; r < 8; r++) O[dt][r] = 0.f;

    const float scale = 0.08838834764831845f;  // 1/sqrt(128)
    const int nkb = 2 * qblk + 2;              // key blocks [0, 2*qblk+1]

    for (int kb = 0; kb < nkb; kb++) {
        // ---- S = scale * Q K^T, causal-masked, 2 n-tiles of 16 keys ----
        float Sm[2][8];
        for (int nt = 0; nt < 2; nt++) {
            v8f s;
            for (int r = 0; r < 8; r++) s[r] = 0.f;
            const unsigned short* kp =
                KVf + (size_t)(b * T_SZ + kb * 32 + nt * 16 + l15) * KVN
                    + hkv * HD + kb8;
            for (int ds = 0; ds < 4; ds++) {
                v16bf kf = load_frag(kp + ds * 32);
                s = __builtin_amdgcn_wmma_f32_16x16x32_bf16(
                        false, qfrag[ds], false, kf, (short)0, s, false, false);
            }
            const int kabs = kb * 32 + nt * 16 + l15;
            for (int r = 0; r < 8; r++) {
                int qabs = q0 + r + 8 * hi;
                float v = s[r] * scale;
                Sm[nt][r] = (kabs <= qabs) ? v : NEG_INF;
            }
        }

        // ---- online softmax (row state replicated per 16-lane half) ----
        float alpha[8];
        for (int r = 0; r < 8; r++) {
            float mx = fmaxf(Sm[0][r], Sm[1][r]);
            for (int m = 1; m < 16; m <<= 1) mx = fmaxf(mx, __shfl_xor(mx, m, 32));
            float mn = fmaxf(mrow[r], mx);
            alpha[r] = __expf(mrow[r] - mn);
            mrow[r]  = mn;
            float p0 = __expf(Sm[0][r] - mn);
            float p1 = __expf(Sm[1][r] - mn);
            Sm[0][r] = p0; Sm[1][r] = p1;
            float ls = p0 + p1;
            for (int m = 1; m < 16; m <<= 1) ls += __shfl_xor(ls, m, 32);
            lrow[r] = lrow[r] * alpha[r] + ls;
        }
        for (int dt = 0; dt < 8; dt++)
            for (int r = 0; r < 8; r++) O[dt][r] *= alpha[r];

        // ---- P: C-layout regs -> LDS bf16 -> A-fragment layout ----
        for (int nt = 0; nt < 2; nt++)
            for (int r = 0; r < 8; r++)
                Plds[w][(r + 8 * hi) * 32 + nt * 16 + l15] = f32_bf16(Sm[nt][r]);
        asm volatile("s_wait_dscnt 0x0" ::: "memory");   // same-wave LDS RAW
        v16bf pfrag;
        {
            union { uint4 u[2]; v16bf v; } f;
            f.u[0] = *reinterpret_cast<const uint4*>(&Plds[w][l15 * 32 + kb8]);
            f.u[1] = *reinterpret_cast<const uint4*>(&Plds[w][l15 * 32 + kb8 + 16]);
            pfrag = f.v;
        }

        // ---- O += P * V  (V^T rows are t-contiguous) ----
        const unsigned short* vbase =
            Vt + (size_t)((b * NKV + hkv) * HD) * T_SZ + kb * 32 + kb8;
        for (int dt = 0; dt < 8; dt++) {
            v16bf vf = load_frag(vbase + (size_t)(dt * 16 + l15) * T_SZ);
            O[dt] = __builtin_amdgcn_wmma_f32_16x16x32_bf16(
                        false, pfrag, false, vf, (short)0, O[dt], false, false);
        }
    }

    // ---- normalize & store y (bf16, [B*T, C] with head-major cols) ----
    float inv[8];
    for (int r = 0; r < 8; r++) inv[r] = 1.0f / lrow[r];
    for (int dt = 0; dt < 8; dt++)
        for (int r = 0; r < 8; r++) {
            int qabs = q0 + r + 8 * hi;
            Yb[(size_t)(b * T_SZ + qabs) * C_SZ + h * HD + dt * 16 + l15] =
                f32_bf16(O[dt][r] * inv[r]);
        }
}

// ---------------------------------------------------------------------------
// Host-side launch
// ---------------------------------------------------------------------------
extern "C" void kernel_launch(void* const* d_in, const int* in_sizes, int n_in,
                              void* d_out, int out_size, void* d_ws, size_t ws_size,
                              hipStream_t stream) {
    (void)in_sizes; (void)n_in; (void)out_size; (void)ws_size;

    const float* x   = (const float*)d_in[0];   // [B,T,C]
    const float* Wq  = (const float*)d_in[1];   // [C,C]
    const float* Wkv = (const float*)d_in[2];   // [C,KVN]
    const float* Wc  = (const float*)d_in[3];   // [C,C]
    float* out = (float*)d_out;                 // [B,T,C] f32

    const size_t M   = (size_t)B_SZ * T_SZ;     // 4096
    const size_t nX  = M * C_SZ;                // 8.39M
    const size_t nWq = (size_t)C_SZ * C_SZ;
    const size_t nWkv= (size_t)C_SZ * KVN;
    const size_t nKV = M * KVN;
    const size_t nVt = (size_t)B_SZ * NKV * HD * T_SZ;  // 2.10M

    char* p = (char*)d_ws;
    unsigned short* xb   = (unsigned short*)p;  p += nX   * 2;
    unsigned short* WqT  = (unsigned short*)p;  p += nWq  * 2;
    unsigned short* WkvT = (unsigned short*)p;  p += nWkv * 2;
    unsigned short* WcT  = (unsigned short*)p;  p += nWq  * 2;
    unsigned short* Qf   = (unsigned short*)p;  p += nX   * 2;
    unsigned short* KVf  = (unsigned short*)p;  p += nKV  * 2;
    unsigned short* Vt   = (unsigned short*)p;  p += nVt  * 2;
    unsigned short* Yb   = (unsigned short*)p;  p += nX   * 2;

    const int TPB = 256;

    // 1) converts / weight transposes
    cvt_f32_bf16_kernel<<<(unsigned)((nX + TPB - 1) / TPB), TPB, 0, stream>>>(x, xb, nX);
    transpose_w_kernel<<<(unsigned)((nWq + TPB - 1) / TPB), TPB, 0, stream>>>(Wq, WqT, C_SZ, C_SZ);
    transpose_w_kernel<<<(unsigned)((nWkv + TPB - 1) / TPB), TPB, 0, stream>>>(Wkv, WkvT, C_SZ, KVN);
    transpose_w_kernel<<<(unsigned)((nWq + TPB - 1) / TPB), TPB, 0, stream>>>(Wc, WcT, C_SZ, C_SZ);

    // 2) projections: Q = x Wq ; KV = x Wkv   (bf16 outputs)
    gemm_bf16_kernel<true><<<dim3(64, 16), 128, 0, stream>>>(xb, WqT,  (void*)Qf,  (int)M, C_SZ, C_SZ);
    gemm_bf16_kernel<true><<<dim3(64,  8), 128, 0, stream>>>(xb, WkvT, (void*)KVf, (int)M, KVN,  C_SZ);

    // 3) V transpose for t-contiguous B fragments
    vt_transpose_kernel<<<(unsigned)((nVt + TPB - 1) / TPB), TPB, 0, stream>>>(KVf, Vt);

    // 4) causal GQA flash attention
    attn_kernel<<<dim3(T_SZ / 64, B_SZ * NH), 128, 0, stream>>>(Qf, KVf, Vt, Yb);

    // 5) output projection: out = y Wc  (f32 output)
    gemm_bf16_kernel<false><<<dim3(64, 16), 128, 0, stream>>>(Yb, WcT, (void*)out, (int)M, C_SZ, C_SZ);
}
